// self_attention_48215302865432
// MI455X (gfx1250) — compile-verified, matrix operand order checked by reading, not run on hardware
//
#include <hip/hip_runtime.h>

// ---------------------------------------------------------------------------
// Self-attention for MI455X (gfx1250, wave32, WMMA bf16 path)
//   x:[4,2048,1024] f32, Wq/Wk/Wv:[1024,1024] f32 (torch-style [out,in])
//   out:[4,2048,1024] f32  (Wu intentionally unused, per reference quirk)
//
// Kernel 1: fused QKV projection (bf16 WMMA), register-pipelined with
//           double-buffered LDS (one barrier per K step). Q/K -> [B,H,S,hd],
//           V -> [B,H,hd,S] (pre-transposed for the PV B-operand).
// Kernel 2: flash attention, 32-key tiles, online softmax. K/V^T tiles
//           double-buffered in LDS via gfx1250 async-to-LDS loads: issue
//           tile t+1, s_wait_asynccnt 4 for tile t, compute overlaps copy.
// ---------------------------------------------------------------------------

typedef __attribute__((ext_vector_type(16))) __bf16 v16bf;
typedef __attribute__((ext_vector_type(8)))  __bf16 v8bf;
typedef __attribute__((ext_vector_type(8)))  float  v8f;
typedef __attribute__((ext_vector_type(4)))  int    v4i;

#define VEC_DIM 1024
#define SEQ     2048
#define BATCH   4
#define HEADS   16
#define HD      64

#if defined(__has_builtin)
#if __has_builtin(__builtin_amdgcn_global_load_async_to_lds_b128)
#define ASYNC_LDS 1
#endif
#endif

static __device__ __forceinline__ v16bf concat8(v8bf lo, v8bf hi) {
  return __builtin_shufflevector(lo, hi, 0,1,2,3,4,5,6,7,8,9,10,11,12,13,14,15);
}

static __device__ __forceinline__ void copy16B_to_lds(__bf16* dst, const __bf16* src) {
#ifdef ASYNC_LDS
  __builtin_amdgcn_global_load_async_to_lds_b128(
      (__attribute__((address_space(1))) v4i*)src,
      (__attribute__((address_space(3))) v4i*)dst, 0, 0);
#else
  *(uint4*)dst = *(const uint4*)src;
#endif
}

static __device__ __forceinline__ void async_wait_n(int keep4) {
#ifdef ASYNC_LDS
#if __has_builtin(__builtin_amdgcn_s_wait_asynccnt)
  if (keep4) __builtin_amdgcn_s_wait_asynccnt(4);
  else       __builtin_amdgcn_s_wait_asynccnt(0);
#else
  if (keep4) asm volatile("s_wait_asynccnt 0x4" ::: "memory");
  else       asm volatile("s_wait_asynccnt 0x0" ::: "memory");
#endif
#endif
}

// ---------------------------------------------------------------------------
// Kernel 1: fused QKV projection.  y = x @ W.T  (bf16 WMMA, f32 accumulate)
// Block: 256 threads (8 waves). Block tile: 128 (M=B*S) x 128 (N = 2 heads).
// Wave tile: 32x64 -> 8 accumulators; double-buffered LDS, reg-pipelined.
// ---------------------------------------------------------------------------
__global__ __launch_bounds__(256)
void qkv_proj_wmma_kernel(const float* __restrict__ x,
                          const float* __restrict__ Wq,
                          const float* __restrict__ Wk,
                          const float* __restrict__ Wv,
                          __bf16* __restrict__ Qo,
                          __bf16* __restrict__ Ko,
                          __bf16* __restrict__ Vo)
{
  const float*  W   = (blockIdx.z == 0) ? Wq : (blockIdx.z == 1) ? Wk : Wv;
  __bf16*       Out = (blockIdx.z == 0) ? Qo : (blockIdx.z == 1) ? Ko : Vo;
  const bool    isV = (blockIdx.z == 2);

  __shared__ __bf16 xs[2][128 * 32];   // x tile, row-major [m][k], bf16
  __shared__ __bf16 wl[2][128 * 32];   // W tile, row-major [n][k], bf16

  const int t    = threadIdx.x;
  const int lane = t & 31;
  const int wv   = t >> 5;          // wave 0..7
  const int wm   = wv & 3;          // M sub-tile (32 rows)
  const int wn   = wv >> 2;         // N sub-tile (64 cols)

  const int m0 = blockIdx.y * 128;  // row into flattened [B*S]
  const int n0 = blockIdx.x * 128;  // output-feature tile (2 heads)

  v8f acc[2][4] = {};

  // staging: each thread converts 16 x-floats and 16 W-floats per K step
  const int srow = t >> 1;          // 0..127
  const int scol = (t & 1) * 16;    // 0 or 16

  const int off0   = (lane < 16) ? 0 : 8;   // 16-bit fragment K-run base
  const int lane15 = lane & 15;

  const float* xrowp = x + (size_t)(m0 + srow) * VEC_DIM + scol;
  const float* wrowp = W + (size_t)(n0 + srow) * VEC_DIM + scol;

  #define LOAD_TILE(dstv, base, k0)                         \
    { const float4* _s = (const float4*)((base) + (k0));    \
      dstv[0] = _s[0]; dstv[1] = _s[1];                     \
      dstv[2] = _s[2]; dstv[3] = _s[3]; }

  #define CVT_STORE(ldsbase, f)                                             \
    { v8bf _h0 = { (__bf16)f[0].x,(__bf16)f[0].y,(__bf16)f[0].z,(__bf16)f[0].w, \
                   (__bf16)f[1].x,(__bf16)f[1].y,(__bf16)f[1].z,(__bf16)f[1].w }; \
      v8bf _h1 = { (__bf16)f[2].x,(__bf16)f[2].y,(__bf16)f[2].z,(__bf16)f[2].w, \
                   (__bf16)f[3].x,(__bf16)f[3].y,(__bf16)f[3].z,(__bf16)f[3].w }; \
      *(v8bf*)((ldsbase) + srow * 32 + scol)     = _h0;                     \
      *(v8bf*)((ldsbase) + srow * 32 + scol + 8) = _h1; }

  { // prologue: stage K-step 0 into buffer 0
    float4 cx[4], cw[4];
    LOAD_TILE(cx, xrowp, 0);
    LOAD_TILE(cw, wrowp, 0);
    CVT_STORE(xs[0], cx);
    CVT_STORE(wl[0], cw);
  }
  __syncthreads();

  int cur = 0;
  for (int k0 = 0; k0 < VEC_DIM; k0 += 32) {
    const bool hn = (k0 + 32) < VEC_DIM;
    float4 nx[4], nw[4];
    if (hn) {               // issue next tile's global loads before compute
      LOAD_TILE(nx, xrowp, k0 + 32);
      LOAD_TILE(nw, wrowp, k0 + 32);
    }

    // fragments + 8 WMMAs from current buffer
    const __bf16* xb = xs[cur];
    const __bf16* wb = wl[cur];
    v16bf a[2], b[4];
    #pragma unroll
    for (int mt = 0; mt < 2; ++mt) {
      const int row = wm * 32 + mt * 16 + lane15;
      v8bf lo = *(const v8bf*)(xb + row * 32 + off0);
      v8bf hi = *(const v8bf*)(xb + row * 32 + off0 + 16);
      a[mt] = concat8(lo, hi);
    }
    #pragma unroll
    for (int nt = 0; nt < 4; ++nt) {
      const int col = wn * 64 + nt * 16 + lane15;
      v8bf lo = *(const v8bf*)(wb + col * 32 + off0);
      v8bf hi = *(const v8bf*)(wb + col * 32 + off0 + 16);
      b[nt] = concat8(lo, hi);
    }
    #pragma unroll
    for (int mt = 0; mt < 2; ++mt)
      #pragma unroll
      for (int nt = 0; nt < 4; ++nt)
        acc[mt][nt] = __builtin_amdgcn_wmma_f32_16x16x32_bf16(
            false, a[mt], false, b[nt], (short)0, acc[mt][nt], false, false);

    if (hn) {               // convert + store next tile into the other buffer
      CVT_STORE(xs[cur ^ 1], nx);
      CVT_STORE(wl[cur ^ 1], nw);
    }
    __syncthreads();        // staging done AND all waves done reading `cur`
    cur ^= 1;
  }
  #undef LOAD_TILE
  #undef CVT_STORE

  // write bf16: Q/K -> [B,H,S,hd]; V -> [B,H,hd,S] (pre-transposed for PV)
  const int base8 = (lane >= 16) ? 8 : 0;
  #pragma unroll
  for (int mt = 0; mt < 2; ++mt) {
    #pragma unroll
    for (int r = 0; r < 8; ++r) {
      const int m  = m0 + wm * 32 + mt * 16 + r + base8;   // flattened B*S row
      const int bb = m >> 11;                              // /2048
      const int s  = m & 2047;
      #pragma unroll
      for (int nt = 0; nt < 4; ++nt) {
        const int ng = n0 + wn * 64 + nt * 16 + lane15;    // 0..1023
        const int h  = ng >> 6;
        const int d  = ng & 63;
        const size_t idx = isV
            ? ((size_t)(bb * HEADS + h) * HD + d) * SEQ + s
            : ((size_t)(bb * HEADS + h) * SEQ + s) * HD + d;
        Out[idx] = (__bf16)acc[mt][nt][r];
      }
    }
  }
}

// ---------------------------------------------------------------------------
// Kernel 2: flash attention per (b,h, 64 query rows). Block: 128 thr (4 waves).
// Wave w owns query rows q0+16w..+15. Key tiles of 32, online softmax.
// Double-buffered async-to-LDS staging: copy of tile t+1 overlaps compute t.
// ---------------------------------------------------------------------------
__global__ __launch_bounds__(128)
void flash_attn_wmma_kernel(const __bf16* __restrict__ Q,
                            const __bf16* __restrict__ K,
                            const __bf16* __restrict__ Vt,
                            float* __restrict__ out)
{
  __shared__ __bf16 ks [2 * 32 * 64];  // K tile row-major [key][hd] (B for QK^T)
  __shared__ __bf16 vtl[2 * 64 * 32];  // V^T tile [hd][key]         (B for PV)
  __shared__ __bf16 ps [4 * 16 * 32];  // per-wave P slab (C-layout -> A-layout)

  const int t      = threadIdx.x;
  const int lane   = t & 31;
  const int wv     = t >> 5;           // wave 0..3
  const int bh     = blockIdx.y;       // 0..63
  const int b      = bh >> 4;
  const int h      = bh & 15;
  const int q0     = blockIdx.x * 64;

  const __bf16* qbase  = Q  + (size_t)bh * SEQ * HD;
  const __bf16* kbase  = K  + (size_t)bh * SEQ * HD;
  const __bf16* vtbase = Vt + (size_t)bh * HD * SEQ;   // [hd][s]

  const int off0   = (lane < 16) ? 0 : 8;
  const int lane15 = lane & 15;
  const int base8  = (lane >= 16) ? 8 : 0;

  // preload Q fragments (A operand for both hd K-steps) into registers
  const int rowq = q0 + wv * 16 + lane15;
  v16bf qf[2];
  #pragma unroll
  for (int kk = 0; kk < 2; ++kk) {
    v8bf lo = *(const v8bf*)(qbase + (size_t)rowq * HD + kk * 32 + off0);
    v8bf hi = *(const v8bf*)(qbase + (size_t)rowq * HD + kk * 32 + off0 + 16);
    qf[kk] = concat8(lo, hi);
  }

  v8f   oacc[4] = {};
  float mrow[8], lrow[8];
  #pragma unroll
  for (int r = 0; r < 8; ++r) { mrow[r] = -3.0e38f; lrow[r] = 0.0f; }

  const float scale = 0.03125f;  // 1/sqrt(1024) — reference's quirk scale

  __bf16* psw = ps + wv * 512;

  // per-thread staging coordinates (4 x 16B async copies per tile)
  const int ke  = t * 16;                 // K tile: 16 bf16 per thread
  const int vd  = t >> 1;                 // V^T tile: row d (two threads/row)
  const int vk0 = (t & 1) * 16;           // 16 keys (32B) per thread

  #define ISSUE_TILE(kt, buf)                                                   \
    { __bf16* _kb = ks  + (buf) * (32 * 64);                                    \
      __bf16* _vb = vtl + (buf) * (64 * 32);                                    \
      copy16B_to_lds(_kb + ke,     kbase + (size_t)(kt) * HD + ke);             \
      copy16B_to_lds(_kb + ke + 8, kbase + (size_t)(kt) * HD + ke + 8);         \
      copy16B_to_lds(_vb + vd * 32 + vk0,                                       \
                     vtbase + (size_t)vd * SEQ + (kt) + vk0);                   \
      copy16B_to_lds(_vb + vd * 32 + vk0 + 8,                                   \
                     vtbase + (size_t)vd * SEQ + (kt) + vk0 + 8); }

  ISSUE_TILE(0, 0);

  int cur = 0;
  for (int kt = 0; kt < SEQ; kt += 32) {
    const bool hn = (kt + 32) < SEQ;
    if (hn) ISSUE_TILE(kt + 32, cur ^ 1);   // overlap copy with compute
    async_wait_n(hn ? 1 : 0);               // tile `kt` landed (in-order)
    if (kt + 64 < SEQ)
      __builtin_prefetch(kbase + (size_t)(kt + 64) * HD + ke, 0, 3);
    __syncthreads();                        // visible to all waves

    const __bf16* ksb = ks  + cur * (32 * 64);
    const __bf16* vtb = vtl + cur * (64 * 32);

    // --- scores S = Q K^T for 16 q-rows x 32 keys ---
    v8f sacc[2] = {};
    #pragma unroll
    for (int nt = 0; nt < 2; ++nt) {
      const int keyc = nt * 16 + lane15;
      #pragma unroll
      for (int kk = 0; kk < 2; ++kk) {
        v8bf lo = *(const v8bf*)(ksb + keyc * 64 + kk * 32 + off0);
        v8bf hi = *(const v8bf*)(ksb + keyc * 64 + kk * 32 + off0 + 16);
        v16bf bf = concat8(lo, hi);
        sacc[nt] = __builtin_amdgcn_wmma_f32_16x16x32_bf16(
            false, qf[kk], false, bf, (short)0, sacc[nt], false, false);
      }
    }

    // --- online softmax (row reductions inside each 16-lane half) ---
    #pragma unroll
    for (int r = 0; r < 8; ++r) {
      float s0 = sacc[0][r] * scale;
      float s1 = sacc[1][r] * scale;
      float tmx = fmaxf(s0, s1);
      #pragma unroll
      for (int msk = 8; msk >= 1; msk >>= 1)
        tmx = fmaxf(tmx, __shfl_xor(tmx, msk, 32));
      const float mnew = fmaxf(mrow[r], tmx);
      const float p0 = __expf(s0 - mnew);
      const float p1 = __expf(s1 - mnew);
      float psum = p0 + p1;
      #pragma unroll
      for (int msk = 8; msk >= 1; msk >>= 1)
        psum += __shfl_xor(psum, msk, 32);
      const float corr = __expf(mrow[r] - mnew);
      lrow[r] = lrow[r] * corr + psum;
      mrow[r] = mnew;
      #pragma unroll
      for (int nt = 0; nt < 4; ++nt) oacc[nt][r] *= corr;
      // C-layout -> LDS (row-major 16x32 P slab, per-wave; DS is in-order)
      psw[(r + base8) * 32 + lane15]      = (__bf16)p0;
      psw[(r + base8) * 32 + 16 + lane15] = (__bf16)p1;
    }

    // --- O += P V  (P as A-operand from LDS, V^T rows as B-operand) ---
    {
      const int rowp = lane15;
      v8bf lo = *(const v8bf*)(psw + rowp * 32 + off0);
      v8bf hi = *(const v8bf*)(psw + rowp * 32 + off0 + 16);
      v16bf pa = concat8(lo, hi);
      #pragma unroll
      for (int nt = 0; nt < 4; ++nt) {
        const int col = nt * 16 + lane15;  // hd column
        v8bf blo = *(const v8bf*)(vtb + col * 32 + off0);
        v8bf bhi = *(const v8bf*)(vtb + col * 32 + off0 + 16);
        v16bf vb2 = concat8(blo, bhi);
        oacc[nt] = __builtin_amdgcn_wmma_f32_16x16x32_bf16(
            false, pa, false, vb2, (short)0, oacc[nt], false, false);
      }
    }

    __syncthreads();   // all waves done reading `cur` before it is re-written
    cur ^= 1;
  }
  #undef ISSUE_TILE

  // normalize and write fp32 output in [B,S,D] layout
  #pragma unroll
  for (int r = 0; r < 8; ++r) {
    const int s   = q0 + wv * 16 + r + base8;
    const float inv = 1.0f / lrow[r];
    #pragma unroll
    for (int nt = 0; nt < 4; ++nt) {
      const int d = nt * 16 + lane15;
      out[((size_t)(b * SEQ + s)) * VEC_DIM + h * HD + d] = oacc[nt][r] * inv;
    }
  }
}

// ---------------------------------------------------------------------------
// Launch
// ---------------------------------------------------------------------------
extern "C" void kernel_launch(void* const* d_in, const int* in_sizes, int n_in,
                              void* d_out, int out_size, void* d_ws, size_t ws_size,
                              hipStream_t stream) {
  const float* x  = (const float*)d_in[0];
  const float* Wq = (const float*)d_in[1];
  const float* Wk = (const float*)d_in[2];
  const float* Wv = (const float*)d_in[3];
  // d_in[4] (Wu) intentionally unused, matching the reference.
  float* out = (float*)d_out;

  const size_t SZ = (size_t)BATCH * SEQ * VEC_DIM;   // 8,388,608 elements
  __bf16* qb = (__bf16*)d_ws;
  __bf16* kb = qb + SZ;
  __bf16* vb = kb + SZ;   // total 48 MB of workspace in bf16 (V stored transposed)

  dim3 gproj(VEC_DIM / 128, (BATCH * SEQ) / 128, 3);  // (8, 64, 3)
  qkv_proj_wmma_kernel<<<gproj, 256, 0, stream>>>(x, Wq, Wk, Wv, qb, kb, vb);

  dim3 gattn(SEQ / 64, BATCH * HEADS);                // (32, 64)
  flash_attn_wmma_kernel<<<gattn, 128, 0, stream>>>(qb, kb, vb, out);
}
